// DeepSeekMoE_83854941487678
// MI455X (gfx1250) — compile-verified, matrix operand order checked by reading, not run on hardware
//
#include <hip/hip_runtime.h>

// ---------------------------------------------------------------------------
// DeepSeek-style MoE (shared + 3 routed experts, top-2) for MI455X (gfx1250).
// bf16 WMMA GEMMs (v_wmma_f32_16x16x32_bf16) with f32 accumulation.
// Tile staging uses gfx1250 async global->LDS loads in GVS addressing mode
// (SGPR base advanced by scalar adds + loop-invariant per-lane offset VGPRs),
// double-buffered LDS, s_wait_asynccnt pipeline waits.
// ---------------------------------------------------------------------------

typedef __attribute__((ext_vector_type(16))) __bf16 v16bf;
typedef __attribute__((ext_vector_type(8)))  float  v8f;
typedef __attribute__((ext_vector_type(4)))  float  v4f;

union Frag16 { v16bf v; v4f f4[2]; };

static constexpr int Hdim = 768;     // hidden
static constexpr int Idim = 1536;    // intermediate
static constexpr int TOK  = 8192;    // B*S

__device__ __forceinline__ unsigned short f2bf(float f) {
    unsigned int b = __float_as_uint(f);
    b += 0x7FFFu + ((b >> 16) & 1u);      // round-to-nearest-even
    return (unsigned short)(b >> 16);
}

// Generic pointers into __shared__ live in the LDS aperture: addr[31:0] is the
// wave-relative LDS byte address (exactly what the async VDST operand wants).
__device__ __forceinline__ unsigned lofs(const void* p) {
    return (unsigned)(unsigned long long)p;
}

// CDNA5 async global->LDS, GVS mode: addr = SADDR(u64) + VADDR(i32).
// ASYNCcnt-tracked (CDNA5 ISA).
__device__ __forceinline__ void async_ld_b128(unsigned ldsOff, unsigned vOff,
                                              unsigned long long sBase) {
    asm volatile("global_load_async_to_lds_b128 %0, %1, %2"
                 :: "v"(ldsOff), "v"(vOff), "s"(sBase)
                 : "memory");
}
#define WAIT_ASYNCCNT(nlit) asm volatile("s_wait_asynccnt " nlit ::: "memory")

// ---------------- convert kernels ----------------
__global__ void cvt_kernel(const float* __restrict__ in,
                           unsigned short* __restrict__ out, int n) {
    int i = blockIdx.x * 256 + threadIdx.x;
    if (i < n) out[i] = f2bf(in[i]);
}

// in: [R][C] f32 row-major  ->  out: [C][R] bf16 row-major (transposed)
__global__ void cvt_t_kernel(const float* __restrict__ in,
                             unsigned short* __restrict__ out, int R, int C) {
    int i = blockIdx.x * 256 + threadIdx.x;
    if (i < R * C) {
        int r = i / C, c = i % C;
        out[(size_t)c * R + r] = f2bf(in[i]);
    }
}

// ---------------- router: one wave per token ----------------
__global__ __launch_bounds__(128)
void router_kernel(const float* __restrict__ x,     // [TOK][H]
                   const float* __restrict__ rw,    // [H][3]
                   const float* __restrict__ rb,    // [3]
                   float* __restrict__ cw4) {       // [TOK][4]
    const int lane  = threadIdx.x & 31;
    const int token = blockIdx.x * 4 + (threadIdx.x >> 5);
    const float* xp = x + (size_t)token * Hdim;

    float s0 = 0.f, s1 = 0.f, s2 = 0.f;
    for (int h = lane; h < Hdim; h += 32) {
        float xv = xp[h];
        s0 += xv * rw[h * 3 + 0];
        s1 += xv * rw[h * 3 + 1];
        s2 += xv * rw[h * 3 + 2];
    }
    #pragma unroll
    for (int off = 16; off > 0; off >>= 1) {
        s0 += __shfl_down(s0, off);
        s1 += __shfl_down(s1, off);
        s2 += __shfl_down(s2, off);
    }
    if (lane == 0) {
        float p[3];
        p[0] = 1.f / (1.f + __expf(-(s0 + rb[0])));
        p[1] = 1.f / (1.f + __expf(-(s1 + rb[1])));
        p[2] = 1.f / (1.f + __expf(-(s2 + rb[2])));
        int   i0 = 0; float m0 = p[0];
        if (p[1] > m0) { m0 = p[1]; i0 = 1; }
        if (p[2] > m0) { m0 = p[2]; i0 = 2; }
        int i1 = -1; float m1 = -1.f;
        #pragma unroll
        for (int e = 0; e < 3; ++e)
            if (e != i0 && p[e] > m1) { m1 = p[e]; i1 = e; }
        float inv = 1.f / (m0 + m1);
        float w[3] = {0.f, 0.f, 0.f};
        w[i0] = m0 * inv;
        w[i1] = m1 * inv;
        float* c = cw4 + (size_t)token * 4;
        c[0] = 1.0f;           // shared expert weight
        c[1] = w[0]; c[2] = w[1]; c[3] = w[2];
    }
}

// ---------------- dual GEMM + SwiGLU ----------------
// h[m,n] = silu(sum_k X[m,k]*G[k,n]) * (sum_k X[m,k]*U[k,n])
// Gt/Ut pre-transposed: [I][H] bf16 (B fragments load like A fragments).
// 128 threads (4 waves), 64x64 block tile, 32x32 wave tile, K-step 32,
// K-loop unrolled by 2 so double-buffer indices are compile-time constants.
__global__ __launch_bounds__(128)
void swiglu_gemm(const unsigned short* __restrict__ X,    // [TOK][H] bf16
                 const unsigned short* __restrict__ Gt,   // [I][H]  bf16
                 const unsigned short* __restrict__ Ut,   // [I][H]  bf16
                 unsigned short* __restrict__ Hout) {     // [TOK][I] bf16
    __shared__ __align__(16) unsigned short As[2][64][32];
    __shared__ __align__(16) unsigned short Bg[2][64][32];
    __shared__ __align__(16) unsigned short Bu[2][64][32];

    const int t    = threadIdx.x;
    const int lane = t & 31;
    const int wave = t >> 5;
    const int wm   = wave >> 1;            // 0..1
    const int wn   = wave & 1;             // 0..1
    const int m_base = blockIdx.x * 64;
    const int n_base = blockIdx.y * 64;

    v8f accg[2][2] = {};
    v8f accu[2][2] = {};

    const int r0 = t >> 2;                 // staging row (per 16B chunk)
    const int c0 = (t & 3) * 8;            // staging col (bf16 elements)

    // loop-invariant per-lane global byte offsets (GVS VADDR operands)
    const unsigned vX0 = (unsigned)(((m_base + r0     ) * Hdim + c0) * 2);
    const unsigned vX1 = (unsigned)(((m_base + r0 + 32) * Hdim + c0) * 2);
    const unsigned vW0 = (unsigned)(((n_base + r0     ) * Hdim + c0) * 2);
    const unsigned vW1 = (unsigned)(((n_base + r0 + 32) * Hdim + c0) * 2);
    // loop-invariant LDS destination offsets, both buffers
    const unsigned lA[2][2] = {{lofs(&As[0][r0][c0]), lofs(&As[0][r0 + 32][c0])},
                               {lofs(&As[1][r0][c0]), lofs(&As[1][r0 + 32][c0])}};
    const unsigned lG[2][2] = {{lofs(&Bg[0][r0][c0]), lofs(&Bg[0][r0 + 32][c0])},
                               {lofs(&Bg[1][r0][c0]), lofs(&Bg[1][r0 + 32][c0])}};
    const unsigned lU[2][2] = {{lofs(&Bu[0][r0][c0]), lofs(&Bu[0][r0 + 32][c0])},
                               {lofs(&Bu[1][r0][c0]), lofs(&Bu[1][r0 + 32][c0])}};

    // issue one K-step's 6 per-thread async chunks into LDS buffer b (b literal!)
    auto stage = [&](int b, int kkv) {
        unsigned long long bx = (unsigned long long)X  + (unsigned long long)(kkv * 2);
        unsigned long long bg = (unsigned long long)Gt + (unsigned long long)(kkv * 2);
        unsigned long long bu = (unsigned long long)Ut + (unsigned long long)(kkv * 2);
        async_ld_b128(lA[b][0], vX0, bx);
        async_ld_b128(lA[b][1], vX1, bx);
        async_ld_b128(lG[b][0], vW0, bg);
        async_ld_b128(lG[b][1], vW1, bg);
        async_ld_b128(lU[b][0], vW0, bu);
        async_ld_b128(lU[b][1], vW1, bu);
    };

    // one K-step of WMMAs from buffer `cur` (cur literal at call sites)
    auto compute = [&](int cur) {
        // ISA 16-bit A/B fragment layout: lanes 0-15 hold K {0..7,16..23},
        // lanes 16-31 hold K {8..15,24..31} for row M (or col N) = lane%16.
        const int koff = (lane < 16) ? 0 : 8;
        const int lr   = lane & 15;
        Frag16 a[2], fg[2], fu[2];
        #pragma unroll
        for (int mt = 0; mt < 2; ++mt) {
            int row = wm * 32 + mt * 16 + lr;
            a[mt].f4[0] = *(const v4f*)&As[cur][row][koff];
            a[mt].f4[1] = *(const v4f*)&As[cur][row][koff + 16];
        }
        #pragma unroll
        for (int nt = 0; nt < 2; ++nt) {
            int row = wn * 32 + nt * 16 + lr;
            fg[nt].f4[0] = *(const v4f*)&Bg[cur][row][koff];
            fg[nt].f4[1] = *(const v4f*)&Bg[cur][row][koff + 16];
            fu[nt].f4[0] = *(const v4f*)&Bu[cur][row][koff];
            fu[nt].f4[1] = *(const v4f*)&Bu[cur][row][koff + 16];
        }
        #pragma unroll
        for (int mt = 0; mt < 2; ++mt)
            #pragma unroll
            for (int nt = 0; nt < 2; ++nt) {
                accg[mt][nt] = __builtin_amdgcn_wmma_f32_16x16x32_bf16(
                    false, a[mt].v, false, fg[nt].v, (short)0, accg[mt][nt], false, false);
                accu[mt][nt] = __builtin_amdgcn_wmma_f32_16x16x32_bf16(
                    false, a[mt].v, false, fu[nt].v, (short)0, accu[mt][nt], false, false);
            }
    };

    stage(0, 0);
    int kk = 0;
    #pragma unroll 1
    for (int it2 = 0; it2 < Hdim / 64; ++it2) {
        // phase 0: consume buf0, prefetch buf1
        if (kk + 32 < Hdim) { stage(1, kk + 32); WAIT_ASYNCCNT("0x6"); }
        else                {                    WAIT_ASYNCCNT("0x0"); }
        __syncthreads();
        compute(0);
        __syncthreads();
        kk += 32;
        // phase 1: consume buf1, prefetch buf0
        if (kk + 32 < Hdim) { stage(0, kk + 32); WAIT_ASYNCCNT("0x6"); }
        else                {                    WAIT_ASYNCCNT("0x0"); }
        __syncthreads();
        compute(1);
        __syncthreads();
        kk += 32;
    }

    // C/D layout: lane holds N = lane%16; VGPR r holds M = r + 8*(lane>=16)
    const int hi = lane >> 4;
    const int lr = lane & 15;
    #pragma unroll
    for (int mt = 0; mt < 2; ++mt)
        #pragma unroll
        for (int nt = 0; nt < 2; ++nt) {
            int col = n_base + wn * 32 + nt * 16 + lr;
            #pragma unroll
            for (int r = 0; r < 8; ++r) {
                int row = m_base + wm * 32 + mt * 16 + hi * 8 + r;
                float g = accg[mt][nt][r];
                float u = accu[mt][nt][r];
                float s = g / (1.0f + __expf(-g));   // silu(g)
                Hout[(size_t)row * Idim + col] = f2bf(s * u);
            }
        }
}

// ---------------- down GEMM + weighted combine ----------------
// out[m,n] (accum? += : =) cw4[m][slot] * sum_k Hin[m,k]*D[k,n]
// Dt pre-transposed: [H][I] bf16. K = Idim.
__global__ __launch_bounds__(128)
void down_gemm(const unsigned short* __restrict__ Hin,   // [TOK][I] bf16
               const unsigned short* __restrict__ Dt,    // [H][I]   bf16
               const float* __restrict__ cw4,            // [TOK][4]
               float* __restrict__ Out,                  // [TOK][H] f32
               int slot, int accum) {
    __shared__ __align__(16) unsigned short As[2][64][32];
    __shared__ __align__(16) unsigned short Bs[2][64][32];

    const int t    = threadIdx.x;
    const int lane = t & 31;
    const int wave = t >> 5;
    const int wm   = wave >> 1;
    const int wn   = wave & 1;
    const int m_base = blockIdx.x * 64;
    const int n_base = blockIdx.y * 64;

    v8f acc[2][2] = {};

    const int r0 = t >> 2;
    const int c0 = (t & 3) * 8;

    const unsigned vA0 = (unsigned)(((m_base + r0     ) * Idim + c0) * 2);
    const unsigned vA1 = (unsigned)(((m_base + r0 + 32) * Idim + c0) * 2);
    const unsigned vB0 = (unsigned)(((n_base + r0     ) * Idim + c0) * 2);
    const unsigned vB1 = (unsigned)(((n_base + r0 + 32) * Idim + c0) * 2);
    const unsigned lA[2][2] = {{lofs(&As[0][r0][c0]), lofs(&As[0][r0 + 32][c0])},
                               {lofs(&As[1][r0][c0]), lofs(&As[1][r0 + 32][c0])}};
    const unsigned lB[2][2] = {{lofs(&Bs[0][r0][c0]), lofs(&Bs[0][r0 + 32][c0])},
                               {lofs(&Bs[1][r0][c0]), lofs(&Bs[1][r0 + 32][c0])}};

    auto stage = [&](int b, int kkv) {
        unsigned long long bh = (unsigned long long)Hin + (unsigned long long)(kkv * 2);
        unsigned long long bd = (unsigned long long)Dt  + (unsigned long long)(kkv * 2);
        async_ld_b128(lA[b][0], vA0, bh);
        async_ld_b128(lA[b][1], vA1, bh);
        async_ld_b128(lB[b][0], vB0, bd);
        async_ld_b128(lB[b][1], vB1, bd);
    };

    auto compute = [&](int cur) {
        const int koff = (lane < 16) ? 0 : 8;
        const int lr   = lane & 15;
        Frag16 a[2], b[2];
        #pragma unroll
        for (int mt = 0; mt < 2; ++mt) {
            int row = wm * 32 + mt * 16 + lr;
            a[mt].f4[0] = *(const v4f*)&As[cur][row][koff];
            a[mt].f4[1] = *(const v4f*)&As[cur][row][koff + 16];
        }
        #pragma unroll
        for (int nt = 0; nt < 2; ++nt) {
            int row = wn * 32 + nt * 16 + lr;
            b[nt].f4[0] = *(const v4f*)&Bs[cur][row][koff];
            b[nt].f4[1] = *(const v4f*)&Bs[cur][row][koff + 16];
        }
        #pragma unroll
        for (int mt = 0; mt < 2; ++mt)
            #pragma unroll
            for (int nt = 0; nt < 2; ++nt)
                acc[mt][nt] = __builtin_amdgcn_wmma_f32_16x16x32_bf16(
                    false, a[mt].v, false, b[nt].v, (short)0, acc[mt][nt], false, false);
    };

    stage(0, 0);
    int kk = 0;
    #pragma unroll 1
    for (int it2 = 0; it2 < Idim / 64; ++it2) {
        if (kk + 32 < Idim) { stage(1, kk + 32); WAIT_ASYNCCNT("0x4"); }
        else                {                    WAIT_ASYNCCNT("0x0"); }
        __syncthreads();
        compute(0);
        __syncthreads();
        kk += 32;
        if (kk + 32 < Idim) { stage(0, kk + 32); WAIT_ASYNCCNT("0x4"); }
        else                {                    WAIT_ASYNCCNT("0x0"); }
        __syncthreads();
        compute(1);
        __syncthreads();
        kk += 32;
    }

    const int hi = lane >> 4;
    const int lr = lane & 15;
    #pragma unroll
    for (int mt = 0; mt < 2; ++mt)
        #pragma unroll
        for (int nt = 0; nt < 2; ++nt) {
            int col = n_base + wn * 32 + nt * 16 + lr;
            #pragma unroll
            for (int r = 0; r < 8; ++r) {
                int row = m_base + wm * 32 + mt * 16 + hi * 8 + r;
                float scale = cw4[(size_t)row * 4 + slot];
                float v = scale * acc[mt][nt][r];
                size_t o = (size_t)row * Hdim + col;
                if (accum) Out[o] += v;
                else       Out[o]  = v;
            }
        }
}

// ---------------- host launch ----------------
extern "C" void kernel_launch(void* const* d_in, const int* in_sizes, int n_in,
                              void* d_out, int out_size, void* d_ws, size_t ws_size,
                              hipStream_t stream) {
    (void)in_sizes; (void)n_in; (void)out_size; (void)ws_size;

    const float* x  = (const float*)d_in[0];   // [4,2048,768]
    const float* sg = (const float*)d_in[1];   // [768,1536]
    const float* su = (const float*)d_in[2];   // [768,1536]
    const float* sd = (const float*)d_in[3];   // [1536,768]
    const float* rg = (const float*)d_in[4];   // [3,768,1536]
    const float* ru = (const float*)d_in[5];   // [3,768,1536]
    const float* rd = (const float*)d_in[6];   // [3,1536,768]
    const float* rw = (const float*)d_in[7];   // [768,3]
    const float* rb = (const float*)d_in[8];   // [3]
    float* out = (float*)d_out;                // [4,2048,768] f32

    // workspace layout
    const size_t WMAT = (size_t)Hdim * Idim;              // 1179648 elems
    char* wsp = (char*)d_ws;
    unsigned short* xbf  = (unsigned short*)wsp;                                  // 12,582,912 B
    unsigned short* wT   = (unsigned short*)(wsp + (size_t)TOK * Hdim * 2);       // 12 mats * 2,359,296 B
    unsigned short* hbuf = (unsigned short*)(wsp + (size_t)TOK * Hdim * 2 + 12 * WMAT * 2);
    float*          cw4  = (float*)(wsp + (size_t)TOK * Hdim * 2 + 12 * WMAT * 2
                                        + (size_t)TOK * Idim * 2);

    auto gT = [&](int s) { return wT + (size_t)s * 3 * WMAT; };
    auto uT = [&](int s) { return wT + (size_t)s * 3 * WMAT + WMAT; };
    auto dT = [&](int s) { return wT + (size_t)s * 3 * WMAT + 2 * WMAT; };

    // f32 -> bf16 conversions (x plain; weights transposed to [N][K])
    {
        int nx = TOK * Hdim;
        cvt_kernel<<<nx / 256, 256, 0, stream>>>(x, xbf, nx);
        int nw = Hdim * Idim;
        int gb = nw / 256;
        cvt_t_kernel<<<gb, 256, 0, stream>>>(sg, gT(0), Hdim, Idim);
        cvt_t_kernel<<<gb, 256, 0, stream>>>(su, uT(0), Hdim, Idim);
        cvt_t_kernel<<<gb, 256, 0, stream>>>(sd, dT(0), Idim, Hdim);
        for (int e = 0; e < 3; ++e) {
            cvt_t_kernel<<<gb, 256, 0, stream>>>(rg + (size_t)e * nw, gT(e + 1), Hdim, Idim);
            cvt_t_kernel<<<gb, 256, 0, stream>>>(ru + (size_t)e * nw, uT(e + 1), Hdim, Idim);
            cvt_t_kernel<<<gb, 256, 0, stream>>>(rd + (size_t)e * nw, dT(e + 1), Idim, Hdim);
        }
    }

    // router combine weights
    router_kernel<<<TOK / 4, 128, 0, stream>>>(x, rw, rb, cw4);

    // experts: slot 0 = shared (weight 1, writes out), slots 1..3 accumulate
    dim3 gridUp(TOK / 64, Idim / 64);   // 128 x 24
    dim3 gridDn(TOK / 64, Hdim / 64);   // 128 x 12
    for (int s = 0; s < 4; ++s) {
        swiglu_gemm<<<gridUp, 128, 0, stream>>>(xbf, gT(s), uT(s), hbuf);
        down_gemm<<<gridDn, 128, 0, stream>>>(hbuf, dT(s), cw4, out, s, s == 0 ? 0 : 1);
    }
}
